// FlashAttentionLayer_83734682402976
// MI455X (gfx1250) — matrix operand order repaired and checked by reading, not compiled, not run on hardware
//
#include <hip/hip_runtime.h>
#include <hip/hip_bf16.h>
#include <math.h>

// ---------------- shapes ----------------
#define BATCH 2
#define NSEQ  2048
#define HID   512
#define NH    8
#define DHD   64
#define ROWS  (BATCH * NSEQ)        // 4096 tokens
#define QKVC  (3 * HID)             // 1536

typedef _Float16 h8  __attribute__((ext_vector_type(8)));
typedef _Float16 h16 __attribute__((ext_vector_type(16)));
typedef float    f8  __attribute__((ext_vector_type(8)));

// ---------------- WMMA helpers ----------------
__device__ __forceinline__ f8 wmma_f16(h16 a, h16 b, f8 c) {
    // D(16x16 f32) = A(16x32 f16) * B(32x16 f16) + C
    return __builtin_amdgcn_wmma_f32_16x16x32_f16(
        /*neg_a=*/false, a, /*neg_b=*/false, b,
        /*c_mod=*/(short)0, c, /*reuse_a=*/false, /*reuse_b=*/false);
}

// A-fragment (16x32, f16). Lane L: row m = L%16, khalf = L/16.
// elements 0..7  <- K = khalf*8 + j          (contiguous run)
// elements 8..15 <- K = 16 + khalf*8 + j     (contiguous run)
// rowp points at &A[row m][kbase]; runs are 16B aligned when kbase%32==0 and
// the row stride is a multiple of 8 halves.
__device__ __forceinline__ h16 load_frag_A(const _Float16* rowp, int khalf) {
    h8 lo = *(const h8*)(rowp + khalf * 8);
    h8 hi = *(const h8*)(rowp + 16 + khalf * 8);
    h16 r;
#pragma unroll
    for (int i = 0; i < 8; ++i) { r[i] = lo[i]; r[i + 8] = hi[i]; }
    return r;
}

// B-fragment (32x16, f16), column-major source (B^T rows contiguous in K).
// Lane L: col n = L%16, khalf = L/16; elements j<8 = B[kbase + khalf*8 + j][n], j>=8 = B[kbase + 16 + khalf*8 + (j-8)][n].
// colp points at &Bt[n][kbase] (Bt = B transposed, so K runs contiguously).
__device__ __forceinline__ h16 load_frag_B(const _Float16* colp, int khalf) {
    h8 lo = *(const h8*)(colp + khalf * 8);
    h8 hi = *(const h8*)(colp + 16 + khalf * 8);
    h16 r;
#pragma unroll
    for (int i = 0; i < 8; ++i) { r[i] = lo[i]; r[i + 8] = hi[i]; }
    return r;
}

// ---------------- conversion kernels ----------------
__global__ void cvt_f32_to_f16(const float* __restrict__ in,
                               _Float16* __restrict__ out, int n) {
    int i = blockIdx.x * blockDim.x + threadIdx.x;
    if (i < n) out[i] = (_Float16)in[i];
}

// out[c][r] = in[r][c]; in is [rows_in][cols_in], out is [cols_in][rows_in]
__global__ void cvt_transpose_f16(const float* __restrict__ in,
                                  _Float16* __restrict__ out,
                                  int rows_in, int cols_in) {
    int i = blockIdx.x * blockDim.x + threadIdx.x;
    if (i < rows_in * cols_in) {
        int c = i / rows_in;
        int r = i - c * rows_in;
        out[(size_t)c * rows_in + r] = (_Float16)in[(size_t)r * cols_in + c];
    }
}

// ---------------- QKV projection ----------------
// Y[4096,1536] = Xh[4096,512] * Wqkv[512,1536] + b ; scatter into
// Q[b,h,n,d], K[b,h,n,d], Vt[b,h,d,n]  (all f16)
__global__ void qkv_gemm(const _Float16* __restrict__ xh,
                         const _Float16* __restrict__ wT,   // [1536][512]
                         const float* __restrict__ bias,    // [1536]
                         _Float16* __restrict__ qb,
                         _Float16* __restrict__ kbuf,
                         _Float16* __restrict__ vT) {
    int wave = (blockIdx.x * blockDim.x + threadIdx.x) >> 5;
    int lane = threadIdx.x & 31;
    const int colTiles = QKVC / 16;            // 96
    int rowTile = wave / colTiles;             // 0..255
    int colTile = wave - rowTile * colTiles;
    int n16 = lane & 15, half = lane >> 4;

    const _Float16* arow = xh + (size_t)(rowTile * 16 + n16) * HID;
    const _Float16* brow = wT + (size_t)(colTile * 16 + n16) * HID;

    f8 acc = {};
    for (int k0 = 0; k0 < HID; k0 += 32)
        acc = wmma_f16(load_frag_A(arow + k0, half),
                       load_frag_B(brow + k0, half), acc);

    int c = colTile * 16 + n16;                // output column (0..1535)
    float bs = bias[c];
    int which = c / HID;                       // 0=q 1=k 2=v (uniform per tile)
    int hid = c - which * HID;
    int hh = hid >> 6, d = hid & 63;
#pragma unroll
    for (int r = 0; r < 8; ++r) {
        int row = rowTile * 16 + r + 8 * half; // token index
        int b = row >> 11, nseq = row & (NSEQ - 1);
        int bh = b * NH + hh;
        _Float16 hv = (_Float16)(acc[r] + bs);
        if (which == 0)      qb  [((size_t)bh * NSEQ + nseq) * DHD + d] = hv;
        else if (which == 1) kbuf[((size_t)bh * NSEQ + nseq) * DHD + d] = hv;
        else                 vT  [((size_t)bh * DHD + d) * NSEQ + nseq] = hv;
    }
}

// ---------------- flash attention ----------------
// One wave = one (b,h, 16-row query tile). Online softmax over 32-key blocks.
__global__ void flash_attn(const _Float16* __restrict__ qb,
                           const _Float16* __restrict__ kbuf,
                           const _Float16* __restrict__ vT,
                           _Float16* __restrict__ aout) { // [4096][512]
    __shared__ _Float16 Plds[4][16 * 32];
    int wv = threadIdx.x >> 5;
    int lane = threadIdx.x & 31;
    int gw = blockIdx.x * 4 + wv;              // 0..2047
    int qtile = gw & 127;                      // 128 q-tiles per (b,h)
    int bh = gw >> 7;                          // 0..15
    int n16 = lane & 15, half = lane >> 4;

    const _Float16* Q = qb   + (size_t)bh * NSEQ * DHD;
    const _Float16* K = kbuf + (size_t)bh * NSEQ * DHD;
    const _Float16* V = vT   + (size_t)bh * DHD * NSEQ;

    // Q A-fragments for d=0..31 and d=32..63 (reused all iterations)
    const _Float16* qrow = Q + (size_t)(qtile * 16 + n16) * DHD;
    h16 qa0 = load_frag_A(qrow, half);
    h16 qa1 = load_frag_A(qrow + 32, half);

    float mrow[8], lrow[8];
    f8 o0 = {}, o1 = {}, o2 = {}, o3 = {};
#pragma unroll
    for (int r = 0; r < 8; ++r) { mrow[r] = -__builtin_inff(); lrow[r] = 0.f; }

    _Float16* P = Plds[wv];
    for (int kb = 0; kb < NSEQ; kb += 32) {
        // ---- S = (Q K^T) for 32 keys: two 16x16 f32 tiles ----
        const _Float16* kr0 = K + (size_t)(kb + n16) * DHD;       // keys kb..+15
        const _Float16* kr1 = K + (size_t)(kb + 16 + n16) * DHD;  // keys +16..+31
        f8 s0 = {}, s1 = {};
        s0 = wmma_f16(qa0, load_frag_B(kr0, half), s0);
        s0 = wmma_f16(qa1, load_frag_B(kr0 + 32, half), s0);
        s1 = wmma_f16(qa0, load_frag_B(kr1, half), s1);
        s1 = wmma_f16(qa1, load_frag_B(kr1 + 32, half), s1);

        // ---- online softmax update (rows live across 16-lane halves) ----
#pragma unroll
        for (int r = 0; r < 8; ++r) {
            float x0 = s0[r] * 0.125f;           // 1/sqrt(64)
            float x1 = s1[r] * 0.125f;
            float mx = fmaxf(x0, x1);
            mx = fmaxf(mx, __shfl_xor(mx, 1, 32));
            mx = fmaxf(mx, __shfl_xor(mx, 2, 32));
            mx = fmaxf(mx, __shfl_xor(mx, 4, 32));
            mx = fmaxf(mx, __shfl_xor(mx, 8, 32));
            float newm = fmaxf(mrow[r], mx);
            float p0 = __expf(x0 - newm);
            float p1 = __expf(x1 - newm);
            float ps = p0 + p1;
            ps += __shfl_xor(ps, 1, 32);
            ps += __shfl_xor(ps, 2, 32);
            ps += __shfl_xor(ps, 4, 32);
            ps += __shfl_xor(ps, 8, 32);
            float cf = __expf(mrow[r] - newm);
            lrow[r] = lrow[r] * cf + ps;
            mrow[r] = newm;
            o0[r] *= cf; o1[r] *= cf; o2[r] *= cf; o3[r] *= cf;
            // stage P (C-layout -> row-major LDS): row m, cols n16 / 16+n16
            int m = r + 8 * half;
            P[m * 32 + n16]      = (_Float16)p0;
            P[m * 32 + 16 + n16] = (_Float16)p1;
        }
        __syncthreads();   // uniform across block; per-wave region RAW guard

        // reload P as an A-fragment (16x32) and do O += P * V
        h16 pa = load_frag_A(P + n16 * 32, half);
        o0 = wmma_f16(pa, load_frag_B(V + (size_t)(n16)      * NSEQ + kb, half), o0);
        o1 = wmma_f16(pa, load_frag_B(V + (size_t)(16 + n16) * NSEQ + kb, half), o1);
        o2 = wmma_f16(pa, load_frag_B(V + (size_t)(32 + n16) * NSEQ + kb, half), o2);
        o3 = wmma_f16(pa, load_frag_B(V + (size_t)(48 + n16) * NSEQ + kb, half), o3);
    }

    // ---- normalize and write [token][h*64 + d] ----
    int b = bh >> 3, hh = bh & 7;
#pragma unroll
    for (int r = 0; r < 8; ++r) {
        int m = r + 8 * half;
        size_t row = (size_t)(b * NSEQ + qtile * 16 + m) * HID + hh * 64;
        float inv = 1.f / lrow[r];
        aout[row + n16]      = (_Float16)(o0[r] * inv);
        aout[row + 16 + n16] = (_Float16)(o1[r] * inv);
        aout[row + 32 + n16] = (_Float16)(o2[r] * inv);
        aout[row + 48 + n16] = (_Float16)(o3[r] * inv);
    }
}

// ---------------- output projection ----------------
__global__ void out_gemm(const _Float16* __restrict__ ah,  // [4096][512]
                         const _Float16* __restrict__ wT,  // [512][512] (out,in)
                         const float* __restrict__ bias,   // [512]
                         float* __restrict__ out) {        // [4096][512] f32
    int wave = (blockIdx.x * blockDim.x + threadIdx.x) >> 5;
    int lane = threadIdx.x & 31;
    const int colTiles = HID / 16;             // 32
    int rowTile = wave / colTiles;
    int colTile = wave - rowTile * colTiles;
    int n16 = lane & 15, half = lane >> 4;

    const _Float16* arow = ah + (size_t)(rowTile * 16 + n16) * HID;
    const _Float16* brow = wT + (size_t)(colTile * 16 + n16) * HID;

    f8 acc = {};
    for (int k0 = 0; k0 < HID; k0 += 32)
        acc = wmma_f16(load_frag_A(arow + k0, half),
                       load_frag_B(brow + k0, half), acc);

    float bs = bias[colTile * 16 + n16];
#pragma unroll
    for (int r = 0; r < 8; ++r) {
        int row = rowTile * 16 + r + 8 * half;
        out[(size_t)row * HID + colTile * 16 + n16] = acc[r] + bs;
    }
}

// ---------------- launch ----------------
extern "C" void kernel_launch(void* const* d_in, const int* in_sizes, int n_in,
                              void* d_out, int out_size, void* d_ws, size_t ws_size,
                              hipStream_t stream) {
    const float* x     = (const float*)d_in[0];   // [2,2048,512]
    const float* w_qkv = (const float*)d_in[1];   // [512,1536]
    const float* b_qkv = (const float*)d_in[2];   // [1536]
    const float* w_out = (const float*)d_in[3];   // [512,512]
    const float* b_out = (const float*)d_in[4];   // [512]
    float* out = (float*)d_out;

    char* ws = (char*)d_ws;
    size_t off = 0;
    _Float16* xh    = (_Float16*)(ws + off); off += (size_t)ROWS * HID * 2;      // 4 MiB
    _Float16* wqkvT = (_Float16*)(ws + off); off += (size_t)QKVC * HID * 2;      // 1.5 MiB
    _Float16* woutT = (_Float16*)(ws + off); off += (size_t)HID * HID * 2;       // 0.5 MiB
    _Float16* qb    = (_Float16*)(ws + off); off += (size_t)BATCH*NH*NSEQ*DHD*2; // 4 MiB
    _Float16* kb    = (_Float16*)(ws + off); off += (size_t)BATCH*NH*NSEQ*DHD*2; // 4 MiB
    _Float16* vT    = (_Float16*)(ws + off); off += (size_t)BATCH*NH*NSEQ*DHD*2; // 4 MiB
    _Float16* aout  = (_Float16*)(ws + off); off += (size_t)ROWS * HID * 2;      // 4 MiB

    // 1) precision conversion / weight transposes
    {
        int n = ROWS * HID;
        cvt_f32_to_f16<<<(n + 255) / 256, 256, 0, stream>>>(x, xh, n);
        int nq = HID * QKVC;
        cvt_transpose_f16<<<(nq + 255) / 256, 256, 0, stream>>>(w_qkv, wqkvT, HID, QKVC);
        int no = HID * HID;
        cvt_transpose_f16<<<(no + 255) / 256, 256, 0, stream>>>(w_out, woutT, HID, HID);
    }
    // 2) QKV projection: (4096/16)*(1536/16)=24576 waves, 4 waves/block
    qkv_gemm<<<24576 / 4, 128, 0, stream>>>(xh, wqkvT, b_qkv, qb, kb, vT);
    // 3) flash attention: 16 (b,h) * 128 q-tiles = 2048 waves
    flash_attn<<<2048 / 4, 128, 0, stream>>>(qb, kb, vT, aout);
    // 4) output projection: (4096/16)*(512/16)=8192 waves
    out_gemm<<<8192 / 4, 128, 0, stream>>>(aout, woutT, b_out, out);
}
